// LSTMEncoder_16045997818339
// MI455X (gfx1250) — matrix operand order repaired and checked by reading, not compiled, not run on hardware
//
#include <hip/hip_runtime.h>
#include <stdint.h>

// ---------------------------------------------------------------------------
// LSTM encoder, fused persistent kernel for MI455X (gfx1250, wave32, WMMA+TDM).
// B=1024, T=336, I=64, H=256.  gates = bias + [x_t | h] @ [W_ih | W_hh]^T
// K = I + H = 320 -> 10 k-steps of v_wmma_f32_16x16x32_bf16 per 16x16 tile.
// Grid: 64 WGs x 512 threads (16 waves = 4/SIMD). Wave w owns h-column tile w
// for each of the 4 gates. Weights are streamed L2 -> LDS by the Tensor Data
// Mover (per-wave 2D descriptors, triple-buffered, TENSORcnt-tracked) so WMMA
// operand fetch is decoupled from L2 latency; the compiler's load-sinking only
// ever exposes ~LDS latency now.
// ---------------------------------------------------------------------------

typedef __attribute__((ext_vector_type(16))) __bf16 v16bf;
typedef __attribute__((ext_vector_type(8)))  __bf16 v8bf;
typedef __attribute__((ext_vector_type(8)))  float  v8f;
typedef uint32_t u32x4 __attribute__((ext_vector_type(4)));
typedef int      i32x8 __attribute__((ext_vector_type(8)));
typedef int      i32x4 __attribute__((ext_vector_type(4)));

#define BATCH 1024
#define HID   256
#define INP   64
#define KTOT  320          // INP + HID
#define NKT   10           // KTOT / 32
#define NT    64           // 4H / 16 column tiles
#define WQ_HALFS (NT * NKT * 512)   // swizzled weight halves (655360 B)

__device__ __forceinline__ uint16_t f2bf_bits(float x) {
    return __builtin_bit_cast(uint16_t, (__bf16)x);   // hw v_cvt, RNE
}
__device__ __forceinline__ float fast_sig(float x) {
    return 1.0f / (1.0f + __expf(-x));
}
__device__ __forceinline__ float fast_tanh(float x) {
    return 2.0f / (1.0f + __expf(-2.0f * x)) - 1.0f;
}
__device__ __forceinline__ v16bf join8(v8bf lo, v8bf hi) {
    return __builtin_shufflevector(lo, hi, 0, 1, 2, 3, 4, 5, 6, 7,
                                   8, 9, 10, 11, 12, 13, 14, 15);
}

// A-operand swizzle (16-bit A-matrix layout, ISA 7.12.2)
__device__ __forceinline__ void a_slot(int m, int K, int& kt, int& lt, int& off) {
    kt = K >> 5;
    int r = K & 31;
    if (r < 8)       { lt = m;      off = r;            }
    else if (r < 16) { lt = m + 16; off = r - 8;        }
    else if (r < 24) { lt = m;      off = 8 + (r - 16); }
    else             { lt = m + 16; off = 8 + (r - 24); }
}

// ---------------------------------------------------------------------------
// Prep: bf16 weights pre-swizzled to per-lane B-fragment layout + fused bias.
// ---------------------------------------------------------------------------
__global__ void lstm_prep_kernel(const float* __restrict__ Wih,
                                 const float* __restrict__ Whh,
                                 const float* __restrict__ bih,
                                 const float* __restrict__ bhh,
                                 uint16_t* __restrict__ wq,
                                 float* __restrict__ bias) {
    int idx = blockIdx.x * 256 + threadIdx.x;           // 0 .. 327679
    if (idx < WQ_HALFS) {
        int i    = idx & 15;
        int lane = (idx >> 4) & 31;
        int ktnt = idx >> 9;            // nt*NKT + kt
        int kt   = ktnt % NKT;
        int nt   = ktnt / NKT;
        int K    = kt * 32 + ((i >> 3) << 4) + ((lane >= 16) ? 8 : 0) + (i & 7);
        int col  = nt * 16 + (lane & 15);
        float v  = (K < INP) ? Wih[col * INP + K]
                             : Whh[col * HID + (K - INP)];
        wq[idx] = f2bf_bits(v);
    }
    if (idx < 4 * HID) bias[idx] = bih[idx] + bhh[idx];
}

// ---------------------------------------------------------------------------
// Main persistent LSTM kernel (512 threads = 16 waves, TDM weight streaming).
// ---------------------------------------------------------------------------
__global__ __launch_bounds__(512, 1)
void lstm_main_kernel(const float* __restrict__ x,
                      const uint16_t* __restrict__ wq,
                      const float* __restrict__ bias,
                      float* __restrict__ out,
                      int T) {
    __shared__ uint16_t A[NKT * 512];                 // 10 KB  [x_t | h]
    __shared__ __align__(16) uint16_t WB[16 * 3 * 2048];  // 192 KB: per-wave 3x4KB

    const int tid  = threadIdx.x;
    const int lane = tid & 31;
    const int w    = tid >> 5;           // wave 0..15
    const int row0 = blockIdx.x * 16;

    // wave-uniform scalars for TDM descriptor construction
    const int ws = __builtin_amdgcn_readfirstlane(w);
    const uint32_t wbBase =
        (uint32_t)(uintptr_t)(&WB[0]) + (uint32_t)(ws * 3) * 4096u;
    const uint64_t gbase =
        (uint64_t)(uintptr_t)wq + (uint64_t)(ws * NKT) * 1024u;

    // Issue one 2D TDM transfer: this wave's 4 B-fragments for k-step kt
    // (4 rows x 1024B, row stride 160KB) -> LDS buffer `buf`.
    auto tdm_issue = [&](int kt, int buf) {
        uint64_t ga = gbase + (uint64_t)kt * 1024u;
        u32x4 g0;
        g0.x = 1u;                                    // count=1, user mode
        g0.y = wbBase + (uint32_t)buf * 4096u;        // lds_addr
        g0.z = (uint32_t)ga;                          // global_addr[31:0]
        g0.w = ((uint32_t)(ga >> 32) & 0x01FFFFFFu)   // global_addr[56:32]
               | 0x80000000u;                         // type=2 (image)
        i32x8 g1;
        g1[0] = (int)(3u << 16);      // data_size=3 (8B elements)
        g1[1] = (int)(128u << 16);    // tensor_dim0 = 128 elements (1KB row)
        g1[2] = (int)(4u << 16);      // tensor_dim1 = 4 rows
        g1[3] = (int)(128u << 16);    // tile_dim0 = 128
        g1[4] = 4;                    // tile_dim1 = 4, tile_dim2 = 0
        g1[5] = 20480;                // tensor_dim0_stride = 160KB / 8B
        g1[6] = 0;
        g1[7] = 0;
        i32x4 z4 = {0, 0, 0, 0};      // groups 2/3 unused (2D tensor)
        i32x8 z8 = {0, 0, 0, 0, 0, 0, 0, 0};
        __builtin_amdgcn_tensor_load_to_lds(g0, g1, z4, z4, z8, 0);
    };

    // zero A (h region must start as zeros)
    for (int i = tid; i < NKT * 512; i += 512) A[i] = 0;

    float bb[4];
#pragma unroll
    for (int g = 0; g < 4; ++g)
        bb[g] = bias[g * HID + w * 16 + (lane & 15)];

    float cc[8], hreg[8];
#pragma unroll
    for (int v = 0; v < 8; ++v) { cc[v] = 0.0f; hreg[v] = 0.0f; }

    // x-tile loader mapping: thread -> (row m, 2 consecutive K); 512 threads
    const int xm  = tid >> 5;
    const int xk2 = (tid & 31) * 2;
    int xkt, xlt, xoff;
    a_slot(xm, xk2, xkt, xlt, xoff);
    const float* xrow = x + ((size_t)(row0 + xm) * T) * INP + xk2;

    __syncthreads();

    for (int t = 0; t < T; ++t) {
        // ---- stage x_t (fp32 -> bf16, swizzled) into LDS --------------------
        {
            const float2 xv = *reinterpret_cast<const float2*>(xrow + (size_t)t * INP);
            uint32_t p = (uint32_t)f2bf_bits(xv.x) | ((uint32_t)f2bf_bits(xv.y) << 16);
            *reinterpret_cast<uint32_t*>(&A[xkt * 512 + xlt * 16 + xoff]) = p;
            if (t + 1 < T)
                __builtin_prefetch(xrow + (size_t)(t + 1) * INP, 0, 1);
        }
        __syncthreads();                // A (x_t + h from prev step) ready

        // WAR guard: previous step's ds reads of WB buffers must be drained
        // before the TDM overwrites them (TDM is unordered vs LDS ops).
        asm volatile("s_wait_dscnt 0x0" ::: "memory");

        // prime the pipeline: slabs for kt=0,1 in flight
        tdm_issue(0, 0);
        tdm_issue(1, 1);

        v8f acc[4];
#pragma unroll
        for (int g = 0; g < 4; ++g) {
            const float bv = bb[g];
            v8f c8 = {bv, bv, bv, bv, bv, bv, bv, bv};
            acc[g] = c8;
        }

#pragma unroll
        for (int kt = 0; kt < NKT; ++kt) {
            // keep 2 TDM transfers in flight; tensor ops complete in order,
            // so waiting <= inflight guarantees slab kt has landed in LDS.
            if (kt <= NKT - 3) {
                tdm_issue(kt + 2, (kt + 2) % 3);
                __builtin_amdgcn_s_wait_tensorcnt(2);
            } else if (kt == NKT - 2) {
                __builtin_amdgcn_s_wait_tensorcnt(1);
            } else {
                __builtin_amdgcn_s_wait_tensorcnt(0);
            }

            const v8bf* ap = reinterpret_cast<const v8bf*>(&A[kt * 512 + lane * 16]);
            v16bf afrag = join8(ap[0], ap[1]);

            const int slab = (w * 3 + (kt % 3)) * 2048 + lane * 16;
            v16bf bf[4];
#pragma unroll
            for (int g = 0; g < 4; ++g) {
                const v8bf* bp =
                    reinterpret_cast<const v8bf*>(&WB[slab + g * 512]);
                bf[g] = join8(bp[0], bp[1]);
            }
#pragma unroll
            for (int g = 0; g < 4; ++g)
                acc[g] = __builtin_amdgcn_wmma_f32_16x16x32_bf16(
                    false, afrag, false, bf[g], (short)0, acc[g], false, false);
        }

        // ---- elementwise LSTM cell update (all 4 gates thread-local) --------
#pragma unroll
        for (int v = 0; v < 8; ++v) {
            float iv = fast_sig(acc[0][v]);
            float fv = fast_sig(acc[1][v]);
            float gv = fast_tanh(acc[2][v]);
            float ov = fast_sig(acc[3][v]);
            float cn = fv * cc[v] + iv * gv;
            cc[v]   = cn;
            hreg[v] = ov * fast_tanh(cn);
        }
        __syncthreads();                // all waves done reading A this step

        // ---- scatter h (bf16) into swizzled A region for next step ----------
#pragma unroll
        for (int v = 0; v < 8; ++v) {
            int m    = v + 8 * (lane >> 4);
            int colh = w * 16 + (lane & 15);
            int kt, lt, off;
            a_slot(m, INP + colh, kt, lt, off);
            A[kt * 512 + lt * 16 + off] = f2bf_bits(hreg[v]);
        }
    }

    // ---- write h_T then c_T (out = [h (B*H) | c (B*H)], fp32) ---------------
#pragma unroll
    for (int v = 0; v < 8; ++v) {
        int m   = v + 8 * (lane >> 4);
        int col = w * 16 + (lane & 15);
        size_t rg = (size_t)(row0 + m) * HID + col;
        out[rg] = hreg[v];
        out[(size_t)BATCH * HID + rg] = cc[v];
    }
}

// ---------------------------------------------------------------------------
extern "C" void kernel_launch(void* const* d_in, const int* in_sizes, int n_in,
                              void* d_out, int out_size, void* d_ws, size_t ws_size,
                              hipStream_t stream) {
    const float* x   = (const float*)d_in[0];
    const float* Wih = (const float*)d_in[1];
    const float* Whh = (const float*)d_in[2];
    const float* bih = (const float*)d_in[3];
    const float* bhh = (const float*)d_in[4];

    const int T = in_sizes[0] / (BATCH * INP);   // 336

    uint16_t* wq  = (uint16_t*)d_ws;
    float* bias   = (float*)((char*)d_ws + (size_t)WQ_HALFS * 2);

    lstm_prep_kernel<<<(WQ_HALFS + 255) / 256, 256, 0, stream>>>(
        Wih, Whh, bih, bhh, wq, bias);

    lstm_main_kernel<<<BATCH / 16, 512, 0, stream>>>(
        x, wq, bias, (float*)d_out, T);
}
